// LSTM_77893526880768
// MI455X (gfx1250) — matrix operand order repaired and checked by reading, not compiled, not run on hardware
//
#include <hip/hip_runtime.h>
#include <hip/hip_bf16.h>
#include <math.h>
#include <stdint.h>

// ---------------------------------------------------------------------------
// LSTM-like scan, B=T=F=H=256, h/c are 256x256 matrices, 255 sequential steps.
// Row-block locality: row m of h_new depends only on row m of xt and h, so a
// block owning a 16-row slab (all 256 columns) never needs inter-block sync.
// 16 blocks x 512 threads (16 wave32s); wave w owns tile (blk rows, cols w*16).
// WMMA f16 -> f32 accumulate; h kept in LDS as f16; c persistent in VGPRs.
// xt staging: global_load_async_to_lds_b128 (CDNA5 async path) from a
// pre-converted f16 copy of x in workspace, with VALU-convert fallback.
// ---------------------------------------------------------------------------

typedef __attribute__((ext_vector_type(16))) _Float16 v16h;
typedef __attribute__((ext_vector_type(8)))  _Float16 v8h;
typedef __attribute__((ext_vector_type(8)))  float    v8f;
typedef __attribute__((ext_vector_type(4)))  float    v4f;

#define HID   256
#define NMAT  8
#define MATSZ (HID * HID)          // 65536 elements per weight matrix
#define LDSTR 264                  // padded LDS row stride (halves): conflict-free ds_load_b128

static __device__ __forceinline__ v8f wmma_f16(v16h a, v16h b, v8f c) {
  // (neg_a, A, neg_b, B, c_mod, C, reuse_a, reuse_b)
  return __builtin_amdgcn_wmma_f32_16x16x32_f16(false, a, false, b, (short)0, c,
                                                false, false);
}

static __device__ __forceinline__ float sigmoid_fast(float x) {
  return 1.0f / (1.0f + __expf(-x));
}
static __device__ __forceinline__ float tanh_fast(float x) {
  return 2.0f / (1.0f + __expf(-2.0f * x)) - 1.0f;   // one v_exp_f32
}

// A-fragment (16x32 f16): lane l -> M = l&15; two 8-half chunks at
// K = k0 + (l>>4)*8 and K = k0 + 16 + (l>>4)*8.
static __device__ __forceinline__ v16h load_a_frag(const _Float16* __restrict__ row_base,
                                                   int k0, int lane) {
  const int m    = lane & 15;
  const int half = lane >> 4;
  const _Float16* r = row_base + m * LDSTR;
  union { v16h v; v8h h[2]; } u;
  u.h[0] = *(const v8h*)(r + k0 + half * 8);
  u.h[1] = *(const v8h*)(r + k0 + 16 + half * 8);
  return u.v;
}

// -------------------------- prep: fp32 -> f16 weights, fuse biases ----------
__global__ __launch_bounds__(256) void lstm_prep_kernel(
    const float* __restrict__ w0, const float* __restrict__ w1,
    const float* __restrict__ w2, const float* __restrict__ w3,
    const float* __restrict__ w4, const float* __restrict__ w5,
    const float* __restrict__ w6, const float* __restrict__ w7,
    const float* __restrict__ bi0, const float* __restrict__ bh0,
    const float* __restrict__ bi1, const float* __restrict__ bh1,
    const float* __restrict__ bi2, const float* __restrict__ bh2,
    const float* __restrict__ bi3, const float* __restrict__ bh3,
    _Float16* __restrict__ w16, float* __restrict__ bias_c) {
  const int idx = blockIdx.x * 256 + threadIdx.x;
  if (idx < NMAT * MATSZ) {
    const float* ws[NMAT] = {w0, w1, w2, w3, w4, w5, w6, w7};
    const int m   = idx >> 16;
    const int off = idx & (MATSZ - 1);
    w16[(m << 16) + (off & (HID - 1)) * HID + (off >> 8)] = (_Float16)ws[m][off];
  } else {
    const int j = idx - NMAT * MATSZ;
    if (j < 4 * HID) {
      const float* bis[4] = {bi0, bi1, bi2, bi3};
      const float* bhs[4] = {bh0, bh1, bh2, bh3};
      bias_c[j] = bis[j >> 8][j & (HID - 1)] + bhs[j >> 8][j & (HID - 1)];
    }
  }
}

// -------------------------- x: fp32 -> f16 full-tensor convert --------------
__global__ __launch_bounds__(256) void x_convert_kernel(
    const float* __restrict__ x, _Float16* __restrict__ x16) {
  const size_t i = ((size_t)blockIdx.x * 256 + threadIdx.x) * 8;
  v4f a = *(const v4f*)(x + i);
  v4f b = *(const v4f*)(x + i + 4);
  v8h h;
  h[0] = (_Float16)a[0]; h[1] = (_Float16)a[1];
  h[2] = (_Float16)a[2]; h[3] = (_Float16)a[3];
  h[4] = (_Float16)b[0]; h[5] = (_Float16)b[1];
  h[6] = (_Float16)b[2]; h[7] = (_Float16)b[3];
  *(v8h*)(x16 + i) = h;
}

// -------------------------- main persistent scan ----------------------------
template <bool USE_ASYNC>
__global__ __launch_bounds__(512) void lstm_scan_kernel(
    const float* __restrict__ x,          // (256,256,256) fp32   (fallback path)
    const _Float16* __restrict__ x16,     // same, f16            (async path)
    const _Float16* __restrict__ w16,     // 8 matrices: ii,hi,if,hf,ig,hg,io,ho
    const float* __restrict__ bias_c,     // 4 x 256 fused biases (i,f,g,o)
    float* __restrict__ out) {            // h (65536 f32) then c (65536 f32)
  __shared__ _Float16 sh_x[16 * LDSTR];
  __shared__ _Float16 sh_h[16 * LDSTR];

  const int tid  = threadIdx.x;
  const int lane = tid & 31;
  const int wv   = tid >> 5;         // 0..15 : column tile, also staged row
  const int blk  = blockIdx.x;       // 0..15 : row block

  for (int i = tid; i < 16 * LDSTR; i += 512) sh_h[i] = (_Float16)0.0f;  // h0 = 0

  v8f creg = {0.f, 0.f, 0.f, 0.f, 0.f, 0.f, 0.f, 0.f};
  v8f hreg = {0.f, 0.f, 0.f, 0.f, 0.f, 0.f, 0.f, 0.f};

  const int ncol  = wv * 16 + (lane & 15);
  const int mbase = (lane >> 4) * 8;

  // ---- staging setup ----
  // async: wave wv stages row wv; lane l moves 16B at column halves l*8..l*8+7
  uint32_t lds_dst = (uint32_t)(uintptr_t)(sh_x + (size_t)wv * LDSTR + lane * 8);
  uint32_t vofs    = ((uint32_t)(blk * 16 + wv) * (uint32_t)MATSZ +
                      (uint32_t)lane * 8u) * 2u;   // byte offset into x16, t=0
  // fallback: thread -> (row tid>>5, 8-float chunk (tid&31)*8)
  const int st_row = tid >> 5;
  const int st_k   = (tid & 31) * 8;
  const float* x_row = x + ((size_t)(blk * 16 + st_row) * HID) * HID + st_k;

  for (int t = 0; t < 255; ++t) {
    // ---- stage x[:, t, :] rows for this block into LDS (f16) ----
    if constexpr (USE_ASYNC) {
      asm volatile("global_load_async_to_lds_b128 %0, %1, %2"
                   :: "v"(lds_dst), "v"(vofs), "s"(x16)
                   : "memory");
      vofs += (uint32_t)HID * 2u;                  // advance t by one slice row
      asm volatile("s_wait_asynccnt 0x0" ::: "memory");
    } else {
      const float* src = x_row + (size_t)t * HID;
      v4f a = *(const v4f*)(src);
      v4f b = *(const v4f*)(src + 4);
      v8h hv;
      hv[0] = (_Float16)a[0]; hv[1] = (_Float16)a[1];
      hv[2] = (_Float16)a[2]; hv[3] = (_Float16)a[3];
      hv[4] = (_Float16)b[0]; hv[5] = (_Float16)b[1];
      hv[6] = (_Float16)b[2]; hv[7] = (_Float16)b[3];
      *(v8h*)(sh_x + st_row * LDSTR + st_k) = hv;
    }
    __syncthreads();   // sh_x ready; previous step's sh_h writes visible

    // ---- 4 gates: acc = bias + xt@Wi + h@Wh over K=256 ----
    v8f acc[4];
#pragma unroll
    for (int g = 0; g < 4; ++g) {
      const float bv = bias_c[g * HID + ncol];
      acc[g] = (v8f){bv, bv, bv, bv, bv, bv, bv, bv};
    }

#pragma unroll 2
    for (int k0 = 0; k0 < HID; k0 += 32) {
      const v16h ax = load_a_frag(sh_x, k0, lane);
      const v16h ah = load_a_frag(sh_h, k0, lane);
      const size_t brow = (size_t)ncol * HID + k0 + (lane >> 4) * 8;  // B: col ncol, K chunks as A
#pragma unroll
      for (int g = 0; g < 4; ++g) {
        union { v16h v; v8h h[2]; } ubi, ubh;
        ubi.h[0] = *(const v8h*)(w16 + (size_t)(2 * g) * MATSZ + brow);
        ubi.h[1] = *(const v8h*)(w16 + (size_t)(2 * g) * MATSZ + brow + 16);
        ubh.h[0] = *(const v8h*)(w16 + (size_t)(2 * g + 1) * MATSZ + brow);
        ubh.h[1] = *(const v8h*)(w16 + (size_t)(2 * g + 1) * MATSZ + brow + 16);
        const v16h bi = ubi.v;
        const v16h bh = ubh.v;
        acc[g] = wmma_f16(ax, bi, acc[g]);
        acc[g] = wmma_f16(ah, bh, acc[g]);
      }
    }

    // ---- elementwise gate math ----
#pragma unroll
    for (int r = 0; r < 8; ++r) {
      const float iv = sigmoid_fast(acc[0][r]);
      const float fv = sigmoid_fast(acc[1][r]);
      const float gv = tanh_fast(acc[2][r]);
      const float ov = sigmoid_fast(acc[3][r]);
      const float cv = fv * creg[r] + iv * gv;
      creg[r] = cv;
      hreg[r] = ov * tanh_fast(cv);
    }

    __syncthreads();   // all reads of sh_h done before overwrite

#pragma unroll
    for (int r = 0; r < 8; ++r) {
      sh_h[(mbase + r) * LDSTR + ncol] = (_Float16)hreg[r];
    }
    // next iteration's first __syncthreads() publishes these writes
  }

  // ---- final output: out[0:65536] = h, out[65536:131072] = c ----
  float* out_h = out;
  float* out_c = out + MATSZ;
#pragma unroll
  for (int r = 0; r < 8; ++r) {
    const int row = blk * 16 + mbase + r;
    out_h[row * HID + ncol] = hreg[r];
    out_c[row * HID + ncol] = creg[r];
  }
}

// ---------------------------------------------------------------------------
extern "C" void kernel_launch(void* const* d_in, const int* in_sizes, int n_in,
                              void* d_out, int out_size, void* d_ws, size_t ws_size,
                              hipStream_t stream) {
  const float* x = (const float*)d_in[0];
  // gate g: iw = d_in[1+4g], ib = d_in[2+4g], hw = d_in[3+4g], hb = d_in[4+4g]

  // workspace layout: [w16: 1MB][biasc: 4KB][pad][x16: 32MB]
  const size_t W16_BYTES = (size_t)NMAT * MATSZ * sizeof(_Float16);   // 1 MB
  const size_t X16_OFF   = W16_BYTES + 8192;                          // 1 MB + 8 KB
  const size_t X_ELEMS   = (size_t)HID * HID * HID;                   // 16.7M
  const size_t NEED      = X16_OFF + X_ELEMS * sizeof(_Float16);

  _Float16* w16   = (_Float16*)d_ws;
  float*    biasc = (float*)((char*)d_ws + W16_BYTES);
  _Float16* x16   = (_Float16*)((char*)d_ws + X16_OFF);

  const int prep_elems = NMAT * MATSZ + 4 * HID;
  lstm_prep_kernel<<<(prep_elems + 255) / 256, 256, 0, stream>>>(
      (const float*)d_in[1],  (const float*)d_in[3],   // ii_w, hi_w
      (const float*)d_in[5],  (const float*)d_in[7],   // if_w, hf_w
      (const float*)d_in[9],  (const float*)d_in[11],  // ig_w, hg_w
      (const float*)d_in[13], (const float*)d_in[15],  // io_w, ho_w
      (const float*)d_in[2],  (const float*)d_in[4],   // ii_b, hi_b
      (const float*)d_in[6],  (const float*)d_in[8],   // if_b, hf_b
      (const float*)d_in[10], (const float*)d_in[12],  // ig_b, hg_b
      (const float*)d_in[14], (const float*)d_in[16],  // io_b, ho_b
      w16, biasc);

  if (ws_size >= NEED) {
    x_convert_kernel<<<(int)(X_ELEMS / 8 / 256), 256, 0, stream>>>(x, x16);
    lstm_scan_kernel<true><<<16, 512, 0, stream>>>(x, x16, w16, biasc,
                                                   (float*)d_out);
  } else {
    lstm_scan_kernel<false><<<16, 512, 0, stream>>>(x, x16, w16, biasc,
                                                    (float*)d_out);
  }
}
